// Spatial_Attention_1666447311424
// MI455X (gfx1250) — matrix operand order repaired and checked by reading, not compiled, not run on hardware
//
#include <hip/hip_runtime.h>

typedef _Float16 half_t;
typedef __attribute__((ext_vector_type(16))) _Float16 v16h;
typedef __attribute__((ext_vector_type(8)))  _Float16 v8h;
typedef __attribute__((ext_vector_type(8)))  float    v8f;

#define NH 6
#define HD 30
#define HDP 32
#define NTOK 64
#define CDIM 180
#define PD 11
#define LN_EPS 1e-5f

// ---------------------------------------------------------------------------
// DPP16 cross-lane permute (VALU, no LDS): butterfly over 16-lane halves
// using xor-span masks {1, 2, 7, 15}.
// ---------------------------------------------------------------------------
template <int CTRL>
__device__ __forceinline__ float dpp_perm_f32(float x) {
  int xi = __builtin_bit_cast(int, x);
  int r  = __builtin_amdgcn_update_dpp(xi, xi, CTRL, 0xf, 0xf, true);
  return __builtin_bit_cast(float, r);
}
#define DPP_QUAD_XOR1  0xB1   // quad_perm [1,0,3,2]
#define DPP_QUAD_XOR2  0x4E   // quad_perm [2,3,0,1]
#define DPP_HALF_MIRR  0x141  // row_half_mirror (xor 7)
#define DPP_ROW_MIRR   0x140  // row_mirror      (xor 15)

__device__ __forceinline__ float red16_max(float x) {
  x = fmaxf(x, dpp_perm_f32<DPP_QUAD_XOR1>(x));
  x = fmaxf(x, dpp_perm_f32<DPP_QUAD_XOR2>(x));
  x = fmaxf(x, dpp_perm_f32<DPP_HALF_MIRR>(x));
  x = fmaxf(x, dpp_perm_f32<DPP_ROW_MIRR>(x));
  return x;
}
__device__ __forceinline__ float red16_sum(float x) {
  x += dpp_perm_f32<DPP_QUAD_XOR1>(x);
  x += dpp_perm_f32<DPP_QUAD_XOR2>(x);
  x += dpp_perm_f32<DPP_HALF_MIRR>(x);
  x += dpp_perm_f32<DPP_ROW_MIRR>(x);
  return x;
}

// ---------------------------------------------------------------------------
// Kernel 1: dynamic relative position bias MLP -> table[225][6] in d_ws
// ---------------------------------------------------------------------------
__global__ void dpb_mlp_kernel(const float* __restrict__ w0, const float* __restrict__ pb,
                               const float* __restrict__ g1, const float* __restrict__ b1,
                               const float* __restrict__ W1, const float* __restrict__ c1,
                               const float* __restrict__ g2, const float* __restrict__ b2,
                               const float* __restrict__ W2, const float* __restrict__ c2,
                               const float* __restrict__ g3, const float* __restrict__ b3,
                               const float* __restrict__ W3, const float* __restrict__ c3,
                               float* __restrict__ table) {
  int r = blockIdx.x * blockDim.x + threadIdx.x;
  if (r >= 225) return;
  float bh = (float)(r / 15 - 7);
  float bw = (float)(r % 15 - 7);
  float x[PD], y[PD];
  #pragma unroll
  for (int j = 0; j < PD; ++j) x[j] = bh * w0[j] + bw * w0[PD + j] + pb[j];

  const float* gs[3] = {g1, g2, g3};
  const float* bs[3] = {b1, b2, b3};
  const float* Ws[3] = {W1, W2, W3};
  const float* cs[3] = {c1, c2, c3};
  #pragma unroll
  for (int st = 0; st < 3; ++st) {
    float m = 0.f;
    #pragma unroll
    for (int j = 0; j < PD; ++j) m += x[j];
    m *= (1.0f / PD);
    float v = 0.f;
    #pragma unroll
    for (int j = 0; j < PD; ++j) { float d = x[j] - m; v += d * d; }
    v *= (1.0f / PD);
    float inv = rsqrtf(v + LN_EPS);
    #pragma unroll
    for (int j = 0; j < PD; ++j) {
      float t = (x[j] - m) * inv * gs[st][j] + bs[st][j];
      y[j] = t > 0.f ? t : 0.f;   // relu
    }
    int ncol = (st == 2) ? NH : PD;
    for (int k = 0; k < ncol; ++k) {
      float acc = cs[st][k];
      #pragma unroll
      for (int j = 0; j < PD; ++j) acc += y[j] * Ws[st][j * ncol + k];
      if (st == 2) table[r * NH + k] = acc;
      else x[k] = acc;
    }
  }
}

// ---------------------------------------------------------------------------
// Kernel 2: windowed attention, one block per 8x8 window, WMMA f16 path
// ---------------------------------------------------------------------------
__global__ __launch_bounds__(256) void win_attn_kernel(
    const float* __restrict__ qkv1, const float* __restrict__ qkv2,
    const float* __restrict__ bias_table, float* __restrict__ out) {
  const long SLICE = 8L * 16384L * CDIM;
  const float* q_g = qkv1;
  const float* v_g = qkv1 + 2 * SLICE;
  const float* k_g = qkv2 + SLICE;

  __shared__ __align__(32) half_t qs[NH][NTOK][HDP];   // [head][token][chan] 24 KB
  __shared__ __align__(32) half_t ks[NH][NTOK][HDP];   // [head][token][chan] 24 KB
  __shared__ __align__(32) half_t vs[NH][HDP][NTOK];   // [head][chan][token] 24 KB
  __shared__ __align__(32) half_t as_[8][16][NTOK];    // per-wave attn strip 16 KB
  __shared__ float bias_l[225 * NH + 2];               // dyn-pos bias table  5.4 KB

  const int w   = blockIdx.x;
  const int b   = w >> 8;
  const int win = w & 255;
  const int wy  = win >> 4, wx = win & 15;
  const long winoff = (long)b * 16384 + (long)wy * 1024 + wx * 8;

  auto growL = [&](int t) -> long {
    return (winoff + ((t >> 3) << 7) + (t & 7)) * (long)CDIM;
  };

  const int tid = threadIdx.x;
  const float scale = 0.18257418583505536f;  // 30^-0.5

  // ---- stage bias table ----
  for (int i = tid; i < 225 * NH; i += 256) bias_l[i] = bias_table[i];

  // ---- cooperative, fully-coalesced Q/K/V staging (f32 -> f16) ----
  for (int i = tid; i < NTOK * CDIM; i += 256) {
    int t = i / CDIM;
    int c = i - t * CDIM;
    int h  = c / HD;
    int kk = c - h * HD;
    long g = growL(t) + c;
    qs[h][t][kk]  = (half_t)(q_g[g] * scale);
    ks[h][t][kk]  = (half_t)k_g[g];
    vs[h][kk][t]  = (half_t)v_g[g];
  }
  // zero the K-padding channels 30,31 (kills the pad contribution in WMMA)
  for (int i = tid; i < NH * NTOK * 2; i += 256) {
    int kk = 30 + (i & 1);
    int t  = (i >> 1) & 63;
    int h  = i >> 7;
    qs[h][t][kk] = (half_t)0.f;
    ks[h][t][kk] = (half_t)0.f;
    vs[h][kk][t] = (half_t)0.f;
  }
  __syncthreads();

  const int lane   = tid & 31;
  const int wave   = tid >> 5;
  const int nhalf  = lane & 15;    // column / row id within 16
  const int hiHalf = lane >> 4;    // 0 or 1

  union V16 { v16h v; v8h h2[2]; half_t e[16]; };

  for (int i = 0; i < 3; ++i) {
    const int u  = wave + 8 * i;   // 0..23
    const int h  = u >> 2;         // head
    const int s  = u & 3;          // 16-row strip
    const int m0 = s * 16;

    // ---- Q A-fragment from LDS ----
    V16 aq;
    {
      const int row = m0 + nhalf;
      const int kb0 = hiHalf * 16;
      aq.h2[0] = *(const v8h*)&qs[h][row][kb0];        // K 0..7 / 16..23
      aq.h2[1] = *(const v8h*)&qs[h][row][kb0 + 8];    // K 8..15 / 24..31
    }

    // ---- attn strip = Q*K^T + bias : bias rides the WMMA C operand ----
    v8f acc[4];
    #pragma unroll
    for (int tn = 0; tn < 4; ++tn) {
      int tok = tn * 16 + nhalf;
      int kb  = hiHalf * 16;
      v16h bk = *(const v16h*)&ks[h][tok][kb];
      v8f c;
      #pragma unroll
      for (int r = 0; r < 8; ++r) {
        int m = m0 + r + 8 * hiHalf;  // query token
        int n = tok;                  // key token
        int idx = ((m >> 3) - (n >> 3) + 7) * 15 + ((m & 7) - (n & 7) + 7);
        c[r] = bias_l[idx * NH + h];
      }
      acc[tn] = __builtin_amdgcn_wmma_f32_16x16x32_f16(false, aq.v, false, bk,
                                                       (short)0, c, false, false);
    }

    // ---- row softmax: 16-lane reductions fully in DPP VALU ----
    #pragma unroll
    for (int r = 0; r < 8; ++r) {
      float mx = fmaxf(fmaxf(acc[0][r], acc[1][r]), fmaxf(acc[2][r], acc[3][r]));
      mx = red16_max(mx);
      float sum = 0.f;
      #pragma unroll
      for (int tn = 0; tn < 4; ++tn) {
        float e = __expf(acc[tn][r] - mx);
        acc[tn][r] = e;
        sum += e;
      }
      sum = red16_sum(sum);
      float inv = 1.0f / sum;
      #pragma unroll
      for (int tn = 0; tn < 4; ++tn) acc[tn][r] *= inv;
    }

    // ---- re-fragment attn via per-wave LDS scratch (C layout -> A layout) ----
    #pragma unroll
    for (int tn = 0; tn < 4; ++tn) {
      #pragma unroll
      for (int r = 0; r < 8; ++r)
        as_[wave][r + 8 * hiHalf][tn * 16 + nhalf] = (half_t)acc[tn][r];
    }
    // wave-local DS RAW -> hardware s_wait_dscnt inserted by compiler

    // ---- P * V : K=64 as two WMMA chunks, 2 output column tiles ----
    v8f z[2] = {{}, {}};
    #pragma unroll
    for (int kc = 0; kc < 2; ++kc) {
      V16 ap;
      int kb0 = kc * 32 + hiHalf * 16;
      ap.h2[0] = *(const v8h*)&as_[wave][nhalf][kb0];
      ap.h2[1] = *(const v8h*)&as_[wave][nhalf][kb0 + 8];
      #pragma unroll
      for (int ct = 0; ct < 2; ++ct) {
        int ch = ct * 16 + nhalf;
        int kb = kc * 32 + hiHalf * 16;
        v16h bv = *(const v16h*)&vs[h][ch][kb];
        z[ct] = __builtin_amdgcn_wmma_f32_16x16x32_f16(false, ap.v, false, bv,
                                                       (short)0, z[ct], false, false);
      }
    }

    // ---- scatter to (B,H,W,C) f32 output ----
    #pragma unroll
    for (int ct = 0; ct < 2; ++ct) {
      int chh = ct * 16 + nhalf;     // channel within head
      if (chh < HD) {
        int cg = h * HD + chh;
        #pragma unroll
        for (int r = 0; r < 8; ++r) {
          int m = m0 + r + 8 * hiHalf;
          out[growL(m) + cg] = z[ct][r];
        }
      }
    }
  }
}

// ---------------------------------------------------------------------------
extern "C" void kernel_launch(void* const* d_in, const int* in_sizes, int n_in,
                              void* d_out, int out_size, void* d_ws, size_t ws_size,
                              hipStream_t stream) {
  const float* qkv1 = (const float*)d_in[0];
  const float* qkv2 = (const float*)d_in[1];
  const float* pos_proj_w = (const float*)d_in[2];
  const float* pos_proj_b = (const float*)d_in[3];
  const float* ln1_g = (const float*)d_in[4];
  const float* ln1_b = (const float*)d_in[5];
  const float* fc1_w = (const float*)d_in[6];
  const float* fc1_b = (const float*)d_in[7];
  const float* ln2_g = (const float*)d_in[8];
  const float* ln2_b = (const float*)d_in[9];
  const float* fc2_w = (const float*)d_in[10];
  const float* fc2_b = (const float*)d_in[11];
  const float* ln3_g = (const float*)d_in[12];
  const float* ln3_b = (const float*)d_in[13];
  const float* fc3_w = (const float*)d_in[14];
  const float* fc3_b = (const float*)d_in[15];
  float* table = (float*)d_ws;                 // 225*6 floats
  float* out   = (float*)d_out;

  dpb_mlp_kernel<<<1, 256, 0, stream>>>(pos_proj_w, pos_proj_b,
                                        ln1_g, ln1_b, fc1_w, fc1_b,
                                        ln2_g, ln2_b, fc2_w, fc2_b,
                                        ln3_g, ln3_b, fc3_w, fc3_b, table);

  // 8 batches * 16*16 windows = 2048 blocks, 256 threads (8 wave32)
  win_attn_kernel<<<2048, 256, 0, stream>>>(qkv1, qkv2, table, out);
}
